// acLSTM_41747082117227
// MI455X (gfx1250) — matrix operand order repaired and checked by reading, not compiled, not run on hardware
//
#include <hip/hip_runtime.h>
#include <stdint.h>

// ---------------------------------------------------------------------------
// acLSTM for MI455X (gfx1250): 3-layer LSTM, T=200 sequential steps.
// Weights converted to bf16 once per call (~44MB -> resident in 192MB L2),
// all GEMMs via V_WMMA_F32_16X16X32_BF16, fp32 cell state, bf16 hidden state.
// Each wave owns a 16x16 hidden block and accumulates its i/f/g/o gate tiles
// so the LSTM pointwise update fuses into the GEMM epilogue.
// ---------------------------------------------------------------------------

typedef __bf16 bf16_t;
typedef __attribute__((ext_vector_type(8)))  __bf16 v8bf;
typedef __attribute__((ext_vector_type(16))) __bf16 v16bf;
typedef __attribute__((ext_vector_type(8)))  float  v8f;

constexpr int Bb = 128;   // batch
constexpr int Tt = 200;   // timesteps
constexpr int Cc = 171;   // frame dim
constexpr int Hh = 1024;  // hidden
constexpr int Kx = 192;   // C padded to multiple of 32 (WMMA K)
constexpr int Nd = 176;   // C padded to multiple of 16 (decoder N)

// ---------------- WMMA helpers (layouts per CDNA5 ISA) ----------

__device__ __forceinline__ v8f wmma_bf16(v16bf a, v16bf b, v8f c) {
  // 8 args: (neg_a, A, neg_b, B, c_mod, C, reuse_a, reuse_b)
  return __builtin_amdgcn_wmma_f32_16x16x32_bf16(false, a, false, b,
                                                 (short)0, c, false, false);
}

// A fragment: 16x32 bf16, A row-major [M, ldk].
// lanes 0-15 : row m, K = {k0..k0+7} in V0-3, {k0+16..k0+23} in V4-7
// lanes 16-31: row m, K = {k0+8..k0+15},     {k0+24..k0+31}
__device__ __forceinline__ v16bf load_a_frag(const bf16_t* __restrict__ A,
                                             int ldk, int mbase, int k0,
                                             int lane) {
  const int m    = mbase + (lane & 15);
  const int koff = (lane >> 4) << 3;  // 0 or 8
  const bf16_t* p = A + (size_t)m * ldk + k0 + koff;
  v8bf lo = *(const v8bf*)(p);
  v8bf hi = *(const v8bf*)(p + 16);
  v16bf a;
#pragma unroll
  for (int i = 0; i < 8; ++i) { a[i] = lo[i]; a[8 + i] = hi[i]; }
  return a;
}

// B fragment: 32x16 bf16 where B[k,n] = W[n,k] (out = act @ W^T), W row-major
// [N, ldk]. lane n (0-15): column n, K = {k0..k0+7}, {k0+16..k0+23};
// lane n+16: K = {k0+8..k0+15}, {k0+24..k0+31} (same order as A fragment).
__device__ __forceinline__ v16bf load_b_frag(const bf16_t* __restrict__ W,
                                             int ldk, int ncol, int k0,
                                             int lane) {
  const bf16_t* p = W + (size_t)ncol * ldk + k0 + ((lane >> 4) << 3);
  v8bf lo = *(const v8bf*)(p);
  v8bf hi = *(const v8bf*)(p + 16);
  v16bf b;
#pragma unroll
  for (int i = 0; i < 8; ++i) { b[i] = lo[i]; b[8 + i] = hi[i]; }
  return b;
}

__device__ __forceinline__ float sigmf(float x) {
  return 1.0f / (1.0f + __expf(-x));
}

// ---------------- prep kernels ---------------------------------------------

// fp32 [N,K] -> bf16 [Np,Kp], zero padded.
__global__ void convert_pad_kernel(const float* __restrict__ src,
                                   bf16_t* __restrict__ dst,
                                   int N, int K, int Np, int Kp) {
  int idx = blockIdx.x * blockDim.x + threadIdx.x;
  int total = Np * Kp;
  if (idx >= total) return;
  int n = idx / Kp, k = idx - n * Kp;
  float v = (n < N && k < K) ? src[(size_t)n * K + k] : 0.0f;
  dst[idx] = (bf16_t)v;
}

__global__ void bias_combine_kernel(const float* __restrict__ a,
                                    const float* __restrict__ b,
                                    float* __restrict__ dst, int N, int Np) {
  int i = blockIdx.x * blockDim.x + threadIdx.x;
  if (i >= Np) return;
  float v = 0.0f;
  if (i < N) { v = a[i]; if (b) v += b[i]; }
  dst[i] = v;
}

__global__ void zero_words_kernel(uint32_t* __restrict__ p, int nwords) {
  int i = blockIdx.x * blockDim.x + threadIdx.x;
  if (i < nwords) p[i] = 0u;
}

// ---------------- LSTM cell kernels ----------------------------------------
// gates[b, n] = A1[b,:] @ W1[n,:] + A2[b,:] @ W2[n,:] + bias[n]
// grid: 64 blocks (j tiles), block: 256 threads = 8 waves (m tiles).
// Each wave accumulates the i/f/g/o 16x16 tiles of one (m,j) hidden block.

__global__ __launch_bounds__(256, 2) void lstm_cell_kernel(
    const bf16_t* __restrict__ A1, int K1, int lda1,
    const bf16_t* __restrict__ W1,            // [4H, lda1]
    const bf16_t* __restrict__ A2,            // h_prev [128, H]
    const bf16_t* __restrict__ W2,            // [4H, H]
    const float* __restrict__ bias,           // [4H] combined b_ih+b_hh
    float* __restrict__ c_state,              // [128, H] in/out (in place)
    bf16_t* __restrict__ h_out) {             // [128, H]
  const int lane  = threadIdx.x & 31;
  const int mbase = (threadIdx.x >> 5) << 4;
  const int jcol  = (blockIdx.x << 4) + (lane & 15);

  v8f ai = {}, af = {}, ag = {}, ao = {};

  for (int k0 = 0; k0 < K1; k0 += 32) {
    v16bf a  = load_a_frag(A1, lda1, mbase, k0, lane);
    v16bf b0 = load_b_frag(W1, lda1, jcol,          k0, lane);
    v16bf b1 = load_b_frag(W1, lda1, jcol + Hh,     k0, lane);
    v16bf b2 = load_b_frag(W1, lda1, jcol + 2 * Hh, k0, lane);
    v16bf b3 = load_b_frag(W1, lda1, jcol + 3 * Hh, k0, lane);
    ai = wmma_bf16(a, b0, ai);
    af = wmma_bf16(a, b1, af);
    ag = wmma_bf16(a, b2, ag);
    ao = wmma_bf16(a, b3, ao);
  }
  for (int k0 = 0; k0 < Hh; k0 += 32) {
    if (k0 + 32 < Hh)  // keep the dominant (L2-resident) weight stream ahead
      __builtin_prefetch(W2 + (size_t)jcol * Hh + k0 + 32, 0, 3);
    v16bf a  = load_a_frag(A2, Hh, mbase, k0, lane);
    v16bf b0 = load_b_frag(W2, Hh, jcol,          k0, lane);
    v16bf b1 = load_b_frag(W2, Hh, jcol + Hh,     k0, lane);
    v16bf b2 = load_b_frag(W2, Hh, jcol + 2 * Hh, k0, lane);
    v16bf b3 = load_b_frag(W2, Hh, jcol + 3 * Hh, k0, lane);
    ai = wmma_bf16(a, b0, ai);
    af = wmma_bf16(a, b1, af);
    ag = wmma_bf16(a, b2, ag);
    ao = wmma_bf16(a, b3, ao);
  }

  // Fused LSTM pointwise epilogue. D layout: elem e -> row mbase+e+8*(lane>=16),
  // col jcol.
  const float bi = bias[jcol];
  const float bf = bias[jcol + Hh];
  const float bg = bias[jcol + 2 * Hh];
  const float bo = bias[jcol + 3 * Hh];
  const int m0 = mbase + ((lane >> 4) << 3);
#pragma unroll
  for (int e = 0; e < 8; ++e) {
    const int idx = (m0 + e) * Hh + jcol;
    float iv = sigmf(ai[e] + bi);
    float fv = sigmf(af[e] + bf);
    float gv = tanhf(ag[e] + bg);
    float ov = sigmf(ao[e] + bo);
    float cn = fv * c_state[idx] + iv * gv;
    c_state[idx] = cn;
    h_out[idx] = (bf16_t)(ov * tanhf(cn));
  }
}

// Cell 1: A1 is built on the fly from the teacher-forcing select
// x = use_gt ? seq[:,t,:] : out[:,t-1,:]  (fp32 -> bf16, padded K=171->192).
__global__ __launch_bounds__(256, 2) void lstm_cell1_kernel(
    const float* __restrict__ seq,            // [B, T, C]
    const float* __restrict__ out_full,       // [B, T, C] (d_out)
    int t, const int* __restrict__ cond_p, const int* __restrict__ gt_p,
    const bf16_t* __restrict__ W1,            // [4H, Kx]
    const bf16_t* __restrict__ A2,            // h1_prev [128, H]
    const bf16_t* __restrict__ W2,            // [4H, H]
    const float* __restrict__ bias,
    float* __restrict__ c_state,
    bf16_t* __restrict__ h_out) {
  const int lane  = threadIdx.x & 31;
  const int mbase = (threadIdx.x >> 5) << 4;
  const int jcol  = (blockIdx.x << 4) + (lane & 15);

  const int gt = *gt_p, cond = *cond_p;
  const int period = gt + cond;
  const bool use_gt = (period > 0) ? ((t % period) < gt) : true;

  const int m = mbase + (lane & 15);           // A-frag row for this lane
  const float* xsrc =
      use_gt ? (seq + ((size_t)m * Tt + t) * Cc)
             : (t > 0 ? (out_full + ((size_t)m * Tt + (t - 1)) * Cc) : nullptr);

  v8f ai = {}, af = {}, ag = {}, ao = {};

  for (int k0 = 0; k0 < Kx; k0 += 32) {
    const int kh = (lane >> 4) << 3;  // 0 or 8
    v16bf a;
#pragma unroll
    for (int i = 0; i < 8; ++i) {
      int ka = k0 + kh + i;
      int kb = k0 + 16 + kh + i;
      float va = (xsrc && ka < Cc) ? xsrc[ka] : 0.0f;
      float vb = (xsrc && kb < Cc) ? xsrc[kb] : 0.0f;
      a[i]     = (bf16_t)va;
      a[8 + i] = (bf16_t)vb;
    }
    v16bf b0 = load_b_frag(W1, Kx, jcol,          k0, lane);
    v16bf b1 = load_b_frag(W1, Kx, jcol + Hh,     k0, lane);
    v16bf b2 = load_b_frag(W1, Kx, jcol + 2 * Hh, k0, lane);
    v16bf b3 = load_b_frag(W1, Kx, jcol + 3 * Hh, k0, lane);
    ai = wmma_bf16(a, b0, ai);
    af = wmma_bf16(a, b1, af);
    ag = wmma_bf16(a, b2, ag);
    ao = wmma_bf16(a, b3, ao);
  }
  for (int k0 = 0; k0 < Hh; k0 += 32) {
    if (k0 + 32 < Hh)
      __builtin_prefetch(W2 + (size_t)jcol * Hh + k0 + 32, 0, 3);
    v16bf a  = load_a_frag(A2, Hh, mbase, k0, lane);
    v16bf b0 = load_b_frag(W2, Hh, jcol,          k0, lane);
    v16bf b1 = load_b_frag(W2, Hh, jcol + Hh,     k0, lane);
    v16bf b2 = load_b_frag(W2, Hh, jcol + 2 * Hh, k0, lane);
    v16bf b3 = load_b_frag(W2, Hh, jcol + 3 * Hh, k0, lane);
    ai = wmma_bf16(a, b0, ai);
    af = wmma_bf16(a, b1, af);
    ag = wmma_bf16(a, b2, ag);
    ao = wmma_bf16(a, b3, ao);
  }

  const float bi = bias[jcol];
  const float bf = bias[jcol + Hh];
  const float bg = bias[jcol + 2 * Hh];
  const float bo = bias[jcol + 3 * Hh];
  const int m0 = mbase + ((lane >> 4) << 3);
#pragma unroll
  for (int e = 0; e < 8; ++e) {
    const int idx = (m0 + e) * Hh + jcol;
    float iv = sigmf(ai[e] + bi);
    float fv = sigmf(af[e] + bf);
    float gv = tanhf(ag[e] + bg);
    float ov = sigmf(ao[e] + bo);
    float cn = fv * c_state[idx] + iv * gv;
    c_state[idx] = cn;
    h_out[idx] = (bf16_t)(ov * tanhf(cn));
  }
}

// Decoder: out[:,t,:] = h3 @ W_dec^T + b_dec. grid: 11 n-tiles, 8 waves = m.
__global__ __launch_bounds__(256) void decoder_kernel(
    const bf16_t* __restrict__ A,     // h3 [128, H]
    const bf16_t* __restrict__ Wd,    // [Nd, H] (rows >= C zero)
    const float* __restrict__ bd,     // [Nd]
    float* __restrict__ out_full, int t) {
  const int lane  = threadIdx.x & 31;
  const int mbase = (threadIdx.x >> 5) << 4;
  const int ncol  = (blockIdx.x << 4) + (lane & 15);

  v8f acc = {};
  for (int k0 = 0; k0 < Hh; k0 += 32) {
    v16bf a = load_a_frag(A, Hh, mbase, k0, lane);
    v16bf b = load_b_frag(Wd, Hh, ncol, k0, lane);
    acc = wmma_bf16(a, b, acc);
  }
  if (ncol < Cc) {
    const float bv = bd[ncol];
    const int m0 = mbase + ((lane >> 4) << 3);
#pragma unroll
    for (int e = 0; e < 8; ++e)
      out_full[((size_t)(m0 + e) * Tt + t) * Cc + ncol] = acc[e] + bv;
  }
}

// ---------------- host-side orchestration ----------------------------------

extern "C" void kernel_launch(void* const* d_in, const int* in_sizes, int n_in,
                              void* d_out, int out_size, void* d_ws,
                              size_t ws_size, hipStream_t stream) {
  const float* seq   = (const float*)d_in[0];
  const float* W_ih1 = (const float*)d_in[1];
  const float* W_hh1 = (const float*)d_in[2];
  const float* b_ih1 = (const float*)d_in[3];
  const float* b_hh1 = (const float*)d_in[4];
  const float* W_ih2 = (const float*)d_in[5];
  const float* W_hh2 = (const float*)d_in[6];
  const float* b_ih2 = (const float*)d_in[7];
  const float* b_hh2 = (const float*)d_in[8];
  const float* W_ih3 = (const float*)d_in[9];
  const float* W_hh3 = (const float*)d_in[10];
  const float* b_ih3 = (const float*)d_in[11];
  const float* b_hh3 = (const float*)d_in[12];
  const float* W_dec = (const float*)d_in[13];
  const float* b_dec = (const float*)d_in[14];
  const int*   cond  = (const int*)d_in[15];
  const int*   gt    = (const int*)d_in[16];
  float* out = (float*)d_out;

  // Workspace carve-out (deterministic; ~45 MB total).
  char* ws = (char*)d_ws;
  size_t off = 0;
  auto carve = [&](size_t bytes) -> char* {
    char* p = ws + off;
    off = (off + bytes + 255) & ~(size_t)255;
    return p;
  };
  bf16_t* Wb_ih1 = (bf16_t*)carve((size_t)4 * Hh * Kx * 2);
  bf16_t* Wb_hh1 = (bf16_t*)carve((size_t)4 * Hh * Hh * 2);
  bf16_t* Wb_ih2 = (bf16_t*)carve((size_t)4 * Hh * Hh * 2);
  bf16_t* Wb_hh2 = (bf16_t*)carve((size_t)4 * Hh * Hh * 2);
  bf16_t* Wb_ih3 = (bf16_t*)carve((size_t)4 * Hh * Hh * 2);
  bf16_t* Wb_hh3 = (bf16_t*)carve((size_t)4 * Hh * Hh * 2);
  bf16_t* Wb_dec = (bf16_t*)carve((size_t)Nd * Hh * 2);
  float* bias1 = (float*)carve(4 * Hh * 4);
  float* bias2 = (float*)carve(4 * Hh * 4);
  float* bias3 = (float*)carve(4 * Hh * 4);
  float* biasd = (float*)carve(Nd * 4);
  // h ping-pong buffers (bf16) then c state (fp32); carved contiguously so a
  // single zero pass covers them.
  char* state0 = ws + off;
  bf16_t* h1[2] = {(bf16_t*)carve((size_t)Bb * Hh * 2),
                   (bf16_t*)carve((size_t)Bb * Hh * 2)};
  bf16_t* h2[2] = {(bf16_t*)carve((size_t)Bb * Hh * 2),
                   (bf16_t*)carve((size_t)Bb * Hh * 2)};
  bf16_t* h3[2] = {(bf16_t*)carve((size_t)Bb * Hh * 2),
                   (bf16_t*)carve((size_t)Bb * Hh * 2)};
  float* c1 = (float*)carve((size_t)Bb * Hh * 4);
  float* c2 = (float*)carve((size_t)Bb * Hh * 4);
  float* c3 = (float*)carve((size_t)Bb * Hh * 4);
  char* state_end = ws + off;
  (void)in_sizes; (void)n_in; (void)out_size; (void)ws_size;

  const int TPB = 256;
  auto blks = [](long long n, int tpb) { return (int)((n + tpb - 1) / tpb); };

  // --- per-call prep: fp32 -> bf16 weight conversion (then L2-resident) ----
  convert_pad_kernel<<<blks((long long)4 * Hh * Kx, TPB), TPB, 0, stream>>>(
      W_ih1, Wb_ih1, 4 * Hh, Cc, 4 * Hh, Kx);
  convert_pad_kernel<<<blks((long long)4 * Hh * Hh, TPB), TPB, 0, stream>>>(
      W_hh1, Wb_hh1, 4 * Hh, Hh, 4 * Hh, Hh);
  convert_pad_kernel<<<blks((long long)4 * Hh * Hh, TPB), TPB, 0, stream>>>(
      W_ih2, Wb_ih2, 4 * Hh, Hh, 4 * Hh, Hh);
  convert_pad_kernel<<<blks((long long)4 * Hh * Hh, TPB), TPB, 0, stream>>>(
      W_hh2, Wb_hh2, 4 * Hh, Hh, 4 * Hh, Hh);
  convert_pad_kernel<<<blks((long long)4 * Hh * Hh, TPB), TPB, 0, stream>>>(
      W_ih3, Wb_ih3, 4 * Hh, Hh, 4 * Hh, Hh);
  convert_pad_kernel<<<blks((long long)4 * Hh * Hh, TPB), TPB, 0, stream>>>(
      W_hh3, Wb_hh3, 4 * Hh, Hh, 4 * Hh, Hh);
  convert_pad_kernel<<<blks((long long)Nd * Hh, TPB), TPB, 0, stream>>>(
      W_dec, Wb_dec, Cc, Hh, Nd, Hh);
  bias_combine_kernel<<<blks(4 * Hh, TPB), TPB, 0, stream>>>(
      b_ih1, b_hh1, bias1, 4 * Hh, 4 * Hh);
  bias_combine_kernel<<<blks(4 * Hh, TPB), TPB, 0, stream>>>(
      b_ih2, b_hh2, bias2, 4 * Hh, 4 * Hh);
  bias_combine_kernel<<<blks(4 * Hh, TPB), TPB, 0, stream>>>(
      b_ih3, b_hh3, bias3, 4 * Hh, 4 * Hh);
  bias_combine_kernel<<<blks(Nd, TPB), TPB, 0, stream>>>(
      b_dec, nullptr, biasd, Cc, Nd);
  {
    int nwords = (int)((state_end - state0) / 4);
    zero_words_kernel<<<blks(nwords, TPB), TPB, 0, stream>>>(
        (uint32_t*)state0, nwords);
  }

  // --- sequential time loop: 4 dependent kernels per step ------------------
  for (int t = 0; t < Tt; ++t) {
    const int p = t & 1, q = p ^ 1;  // prev / next h buffers
    lstm_cell1_kernel<<<Hh / 16, TPB, 0, stream>>>(
        seq, out, t, cond, gt, Wb_ih1, h1[p], Wb_hh1, bias1, c1, h1[q]);
    lstm_cell_kernel<<<Hh / 16, TPB, 0, stream>>>(
        h1[q], Hh, Hh, Wb_ih2, h2[p], Wb_hh2, bias2, c2, h2[q]);
    lstm_cell_kernel<<<Hh / 16, TPB, 0, stream>>>(
        h2[q], Hh, Hh, Wb_ih3, h3[p], Wb_hh3, bias3, c3, h3[q]);
    decoder_kernel<<<Nd / 16, TPB, 0, stream>>>(h3[q], Wb_dec, biasd, out, t);
  }
}
